// BatchGGNNEncoder_22325240004845
// MI455X (gfx1250) — matrix operand order repaired and checked by reading, not compiled, not run on hardware
//
#include <hip/hip_runtime.h>
#include <hip/hip_bf16.h>

// ---- problem constants (match reference) ----
#define Bb    32
#define MAXNN 1024
#define Ff    215
#define Hd    256
#define Tt    8
#define Ll    3
#define Nn    (Bb * MAXNN)      // 32768
#define Ee    524288
#define FP    224               // F padded to multiple of 32 for WMMA K
#define K2    (Tt * Hd)         // 2048: K of the aggregated message GEMM
#define G3    (3 * Hd)          // 768: GRU gate width

typedef __bf16 v16bf __attribute__((ext_vector_type(16)));
typedef __bf16 v8bf  __attribute__((ext_vector_type(8)));
typedef float  v8f   __attribute__((ext_vector_type(8)));
typedef unsigned int u32x4 __attribute__((ext_vector_type(4)));
typedef int          i32x4 __attribute__((ext_vector_type(4)));
typedef int          i32x8 __attribute__((ext_vector_type(8)));

// ---- compile-time capability probes (text shows up in compiler stderr) ----
#if __has_builtin(__builtin_amdgcn_tensor_load_to_lds)
#define HAVE_TDM 1
#pragma message("CDNA5-PROBE: tensor_load_to_lds builtin AVAILABLE")
#if __has_include(<hip/amd_detail/amd_gfx1250_TDM.h>)
#define TDM_6ARG 1
#pragma message("CDNA5-PROBE: TDM 6-arg form (therock headers present)")
#else
#pragma message("CDNA5-PROBE: TDM 5-arg form (no therock TDM header)")
#endif
#else
#pragma message("CDNA5-PROBE: tensor_load_to_lds builtin MISSING -> fallback copies")
#endif
#if __has_builtin(__builtin_amdgcn_global_load_async_to_lds_b128)
#pragma message("CDNA5-PROBE: global_load_async_to_lds_b128 builtin AVAILABLE")
#endif
#if __has_builtin(__builtin_amdgcn_s_wait_asynccnt)
#pragma message("CDNA5-PROBE: s_wait_asynccnt builtin AVAILABLE")
#endif
#if __has_builtin(__builtin_amdgcn_global_atomic_fadd_v2bf16)
#pragma message("CDNA5-PROBE: global_atomic_fadd_v2bf16 builtin AVAILABLE")
#endif

__device__ __forceinline__ void tdm_wait0() {
#if __has_builtin(__builtin_amdgcn_s_wait_tensorcnt)
  __builtin_amdgcn_s_wait_tensorcnt(0);
#else
  asm volatile("s_wait_tensorcnt 0x0" ::: "memory");
#endif
}
__device__ __forceinline__ void tdm_wait1() {
#if __has_builtin(__builtin_amdgcn_s_wait_tensorcnt)
  __builtin_amdgcn_s_wait_tensorcnt(1);
#else
  asm volatile("s_wait_tensorcnt 0x1" ::: "memory");
#endif
}

#ifdef HAVE_TDM
// 2-D TDM copy: tile_y rows of tile_x 2-byte elements, global row stride
// stride_elems, packed contiguously into LDS at lds_off. tile_y==1 => 1-D.
__device__ __forceinline__ void tdm_load_2d(unsigned lds_off,
                                            unsigned long long gaddr,
                                            unsigned tile_x, unsigned tile_y,
                                            unsigned stride_elems) {
  u32x4 g0;
  g0[0] = 1u;                                       // count=1, user descriptor
  g0[1] = lds_off;                                  // lds_addr (bytes)
  g0[2] = (unsigned)(gaddr & 0xffffffffu);          // global_addr[31:0]
  g0[3] = (unsigned)((gaddr >> 32) & 0x01ffffffu)   // global_addr[56:32]
        | (2u << 30);                               // type = 2 ("image")
  i32x8 g1;
  g1[0] = (1 << 16);                                // data_size=1 (2 bytes)
  g1[1] = (int)((stride_elems & 0xffffu) << 16);    // tensor_dim0[15:0]
  g1[2] = (int)(((stride_elems >> 16) & 0xffffu)    // tensor_dim0[31:16]
        | ((tile_y & 0xffffu) << 16));              // tensor_dim1[15:0]
  g1[3] = (int)((tile_x & 0xffffu) << 16);          // tile_dim0
  g1[4] = (int)(tile_y & 0xffffu);                  // tile_dim1
  g1[5] = (int)stride_elems;                        // tensor_dim0_stride[31:0]
  g1[6] = 0;
  g1[7] = 0;
  i32x4 z4 = {0, 0, 0, 0};
#ifdef TDM_6ARG
  i32x8 z8 = {0, 0, 0, 0, 0, 0, 0, 0};
  __builtin_amdgcn_tensor_load_to_lds(g0, g1, z4, z4, z8, 0);
#else
  __builtin_amdgcn_tensor_load_to_lds(g0, g1, z4, z4, 0);
#endif
}
#endif

// Packed bf16x2 atomic add (GLOBAL_ATOMIC_PK_ADD_BF16, no return).
__device__ __forceinline__ void atomic_pk_add_bf16(__bf16* addr, unsigned packed) {
  asm volatile("global_atomic_pk_add_bf16 %0, %1, off"
               :: "v"((unsigned long long)(uintptr_t)addr), "v"(packed)
               : "memory");
}

__device__ __forceinline__ v8f vzero8() {
  v8f v;
#pragma unroll
  for (int i = 0; i < 8; ++i) v[i] = 0.0f;
  return v;
}

__device__ __forceinline__ v8f wmma_bf16(v16bf a, v16bf b, v8f c) {
  return __builtin_amdgcn_wmma_f32_16x16x32_bf16(false, a, false, b, (short)0, c,
                                                 false, false);
}

// A-matrix fragment (16x32 bf16, row-major, leading dim ld). ISA layout:
// lane<16: row=lane, K=kk+{0..7,16..23}; lane>=16: row=lane-16, K=kk+{8..15,24..31}.
__device__ __forceinline__ v16bf afrag_bf(const __bf16* base, int ld, int row0,
                                          int kk, int lane) {
  int r  = lane & 15;
  int kh = (lane >> 4) << 3;
  const __bf16* p = base + (size_t)(row0 + r) * ld + kk + kh;
  v8bf lo = *(const v8bf*)p;
  v8bf hi = *(const v8bf*)(p + 16);
  return __builtin_shufflevector(lo, hi, 0, 1, 2, 3, 4, 5, 6, 7,
                                 8, 9, 10, 11, 12, 13, 14, 15);
}

// B-matrix fragment (32x16): weights row-major (out x in); lane<16: col=lane,
// K=kk+{0..7,16..23}; lane>=16: col=lane-16, K=kk+{8..15,24..31}.
__device__ __forceinline__ v16bf bfrag(const __bf16* W, int ld, int col0,
                                       int kk, int lane) {
  int c  = lane & 15;
  int kh = (lane >> 4) << 3;
  const __bf16* p = W + (size_t)(col0 + c) * ld + kk + kh;
  v8bf lo = *(const v8bf*)p;
  v8bf hi = *(const v8bf*)(p + 16);
  return __builtin_shufflevector(lo, hi, 0, 1, 2, 3, 4, 5, 6, 7,
                                 8, 9, 10, 11, 12, 13, 14, 15);
}

__device__ __forceinline__ float sigm(float x) { return 1.0f / (1.0f + __expf(-x)); }
__device__ __forceinline__ float tanh_fast(float x) {
  float e = __expf(-2.0f * x);
  return (1.0f - e) / (1.0f + e);
}

// ---------------- conversion / packing kernels ----------------

__global__ void k_conv_x(const float* __restrict__ X, __bf16* __restrict__ Xb) {
  for (int i = blockIdx.x * blockDim.x + threadIdx.x; i < Nn * FP;
       i += gridDim.x * blockDim.x) {
    int n = i / FP, k = i - n * FP;
    Xb[i] = (k < Ff) ? (__bf16)X[(size_t)n * Ff + k] : (__bf16)0.0f;
  }
}

__global__ void k_conv_wp(const float* __restrict__ Wp, __bf16* __restrict__ Wpb) {
  for (int i = blockIdx.x * blockDim.x + threadIdx.x; i < Hd * FP;
       i += gridDim.x * blockDim.x) {
    int e = i / FP, k = i - e * FP;
    Wpb[i] = (k < Ff) ? (__bf16)Wp[(size_t)e * Ff + k] : (__bf16)0.0f;
  }
}

// Wm (L,T,Hout,Hin) fp32 -> Wmb (L, Hout, T*Hin) bf16: one K=2048 GEMM.
__global__ void k_conv_wm(const float* __restrict__ Wm, __bf16* __restrict__ Wmb) {
  const int total = Ll * Tt * Hd * Hd;
  for (int i = blockIdx.x * blockDim.x + threadIdx.x; i < total;
       i += gridDim.x * blockDim.x) {
    int l = i / (Tt * Hd * Hd);
    int r = i - l * (Tt * Hd * Hd);
    int t = r / (Hd * Hd);
    int r2 = r - t * (Hd * Hd);
    int e = r2 / Hd;
    int d = r2 - e * Hd;
    Wmb[(size_t)l * Hd * K2 + (size_t)e * K2 + t * Hd + d] = (__bf16)Wm[i];
  }
}

__global__ void k_conv_w2(const float* __restrict__ Wih,
                          const float* __restrict__ Whh,
                          __bf16* __restrict__ Wihb, __bf16* __restrict__ Whhb) {
  const int total = Ll * G3 * Hd;
  for (int i = blockIdx.x * blockDim.x + threadIdx.x; i < total;
       i += gridDim.x * blockDim.x) {
    Wihb[i] = (__bf16)Wih[i];
    Whhb[i] = (__bf16)Whh[i];
  }
}

// ---------------- input projection: h = X @ Wp.T + bp ----------------
__global__ void k_proj(const __bf16* __restrict__ Xb, const __bf16* __restrict__ Wpb,
                       const float* __restrict__ bp, float* __restrict__ h,
                       __bf16* __restrict__ hb) {
  int lane = threadIdx.x & 31;
  int tile = blockIdx.x * (blockDim.x >> 5) + (threadIdx.x >> 5);
  int row0 = (tile >> 4) << 4;
  int col0 = (tile & 15) << 4;
  v8f acc = vzero8();
#pragma unroll
  for (int kk = 0; kk < FP; kk += 32)
    acc = wmma_bf16(afrag_bf(Xb, FP, row0, kk, lane),
                    bfrag(Wpb, FP, col0, kk, lane), acc);
  int c = col0 + (lane & 15);
  float bias = bp[c];
#pragma unroll
  for (int j = 0; j < 8; ++j) {
    int rr = row0 + j + ((lane >> 4) << 3);
    float v = acc[j] + bias;
    size_t o = (size_t)rr * Hd + c;
    h[o]  = v;
    hb[o] = (__bf16)v;
  }
}

// ------- per-edge scatter: A[dst, type, :] += hb[src, :]  (bf16, pk atomics) -------
__global__ void k_edge(const __bf16* __restrict__ hb, const int* __restrict__ src,
                       const int* __restrict__ dst, const int* __restrict__ et,
                       __bf16* __restrict__ A, int* __restrict__ cnt) {
  int e = blockIdx.x * (blockDim.x >> 5) + (threadIdx.x >> 5);
  if (e >= Ee) return;
  int lane = threadIdx.x & 31;
  int s = src[e], d = dst[e], t = et[e];
  uint4 x = *(const uint4*)(hb + (size_t)s * Hd + lane * 8);
  __bf16* ad = A + (size_t)d * K2 + t * Hd + lane * 8;
  atomic_pk_add_bf16(ad + 0, x.x);
  atomic_pk_add_bf16(ad + 2, x.y);
  atomic_pk_add_bf16(ad + 4, x.z);
  atomic_pk_add_bf16(ad + 6, x.w);
  if (lane == 0)
    __hip_atomic_fetch_add(cnt + (size_t)d * Tt + t, 1, __ATOMIC_RELAXED,
                           __HIP_MEMORY_SCOPE_AGENT);
}

// ---- message GEMM: m = A(N,2048) @ Wmb[l].T + Σ_t cnt[n,t]·bm[t] ----
// Block = one 16-row band. The 64KB A-panel is split into two K-halves loaded
// by two TDM descriptors issued back-to-back; compute on half 0 overlaps the
// DMA of half 1 (s_wait_tensorcnt 1 -> 0, in-order TDM completion).
__global__ void k_msg(const __bf16* __restrict__ A, const __bf16* __restrict__ Wmb,
                      const int* __restrict__ cnt, const float* __restrict__ bm,
                      __bf16* __restrict__ mb) {
  __shared__ __align__(32) __bf16 As[2][16 * 1024];   // 2 x 32KB
  int lane = threadIdx.x & 31;
  int w    = threadIdx.x >> 5;
  int row0 = blockIdx.x << 4;
  const __bf16* gbase = A + (size_t)row0 * K2;
#ifdef HAVE_TDM
  if (w == 0) {
    tdm_load_2d((unsigned)(uintptr_t)(void*)As[0],
                (unsigned long long)(uintptr_t)gbase, 1024, 16, K2);
    tdm_load_2d((unsigned)(uintptr_t)(void*)As[1],
                (unsigned long long)(uintptr_t)(gbase + 1024), 1024, 16, K2);
  }
#else
  for (int i = threadIdx.x; i < 16 * 1024 / 8; i += 256) {
    int rowi = i >> 7, coli = i & 127;             // 128 uint4 per row-half
    ((uint4*)As[0])[i] = *(const uint4*)(gbase + (size_t)rowi * K2 + coli * 8);
    ((uint4*)As[1])[i] = *(const uint4*)(gbase + (size_t)rowi * K2 + 1024 + coli * 8);
  }
  __syncthreads();
#endif
  int c0 = w << 5;                                  // wave owns col-tiles 2w,2w+1
  v8f acc[2] = {vzero8(), vzero8()};
#pragma unroll
  for (int half = 0; half < 2; ++half) {
#ifdef HAVE_TDM
    if (w == 0) { if (half == 0) tdm_wait1(); else tdm_wait0(); }
    __syncthreads();
#endif
    for (int kk = 0; kk < 1024; kk += 32) {
      v16bf a = afrag_bf(As[half], 1024, 0, kk, lane);   // ds_load_b128 x2
      acc[0] = wmma_bf16(a, bfrag(Wmb, K2, c0, half * 1024 + kk, lane), acc[0]);
      acc[1] = wmma_bf16(a, bfrag(Wmb, K2, c0 + 16, half * 1024 + kk, lane), acc[1]);
    }
  }
#pragma unroll
  for (int tt = 0; tt < 2; ++tt) {
    int c = c0 + tt * 16 + (lane & 15);
    float bmv[Tt];
#pragma unroll
    for (int t = 0; t < Tt; ++t) bmv[t] = bm[t * Hd + c];
#pragma unroll
    for (int j = 0; j < 8; ++j) {
      int rr = row0 + j + ((lane >> 4) << 3);
      float bias = 0.0f;
#pragma unroll
      for (int t = 0; t < Tt; ++t) bias += (float)cnt[(size_t)rr * Tt + t] * bmv[t];
      mb[(size_t)rr * Hd + c] = (__bf16)(acc[tt][j] + bias);
    }
  }
}

// -------- fused GRU: LDS-staged band operands + register-resident gates --------
// All 8 waves consume identical A-fragments, so stage the 8KB mb/hb bands into
// LDS once (TDM, contiguous 1-D tiles) instead of 8x redundant global reads.
__global__ void k_gru(const __bf16* __restrict__ mb, __bf16* __restrict__ hb,
                      float* __restrict__ h, const __bf16* __restrict__ Wihb,
                      const __bf16* __restrict__ Whhb,
                      const float* __restrict__ bih, const float* __restrict__ bhh) {
  __shared__ __align__(32) __bf16 Ms[16 * Hd];   // 8KB
  __shared__ __align__(32) __bf16 Hs[16 * Hd];   // 8KB
  int lane = threadIdx.x & 31;
  int w    = threadIdx.x >> 5;
  int row0 = blockIdx.x << 4;
#ifdef HAVE_TDM
  if (w == 0) {
    tdm_load_2d((unsigned)(uintptr_t)(void*)Ms,
                (unsigned long long)(uintptr_t)(mb + (size_t)row0 * Hd),
                16 * Hd, 1, 16 * Hd);
    tdm_load_2d((unsigned)(uintptr_t)(void*)Hs,
                (unsigned long long)(uintptr_t)(hb + (size_t)row0 * Hd),
                16 * Hd, 1, 16 * Hd);
    tdm_wait0();
  }
  __syncthreads();
#else
  for (int i = threadIdx.x; i < 16 * Hd / 8; i += 256) {
    ((uint4*)Ms)[i] = ((const uint4*)(mb + (size_t)row0 * Hd))[i];
    ((uint4*)Hs)[i] = ((const uint4*)(hb + (size_t)row0 * Hd))[i];
  }
  __syncthreads();
#endif
  int colBase = w << 5;
  v8f accI[6], accH[6];
#pragma unroll
  for (int i = 0; i < 6; ++i) { accI[i] = vzero8(); accH[i] = vzero8(); }
#pragma unroll
  for (int kk = 0; kk < Hd; kk += 32) {
    v16bf am = afrag_bf(Ms, Hd, 0, kk, lane);
    v16bf ah = afrag_bf(Hs, Hd, 0, kk, lane);
#pragma unroll
    for (int g = 0; g < 3; ++g) {
#pragma unroll
      for (int t = 0; t < 2; ++t) {
        int bc = g * Hd + colBase + t * 16;
        accI[g * 2 + t] = wmma_bf16(am, bfrag(Wihb, Hd, bc, kk, lane), accI[g * 2 + t]);
        accH[g * 2 + t] = wmma_bf16(ah, bfrag(Whhb, Hd, bc, kk, lane), accH[g * 2 + t]);
      }
    }
  }
#pragma unroll
  for (int t = 0; t < 2; ++t) {
    int c = colBase + t * 16 + (lane & 15);
    float biR = bih[c], biZ = bih[Hd + c], biN = bih[2 * Hd + c];
    float bhR = bhh[c], bhZ = bhh[Hd + c], bhN = bhh[2 * Hd + c];
#pragma unroll
    for (int j = 0; j < 8; ++j) {
      int rr = row0 + j + ((lane >> 4) << 3);
      float rg = sigm(accI[0 + t][j] + biR + accH[0 + t][j] + bhR);
      float zg = sigm(accI[2 + t][j] + biZ + accH[2 + t][j] + bhZ);
      float ng = tanh_fast(accI[4 + t][j] + biN + rg * (accH[4 + t][j] + bhN));
      size_t o = (size_t)rr * Hd + c;
      float hold = h[o];
      float hnew = (1.0f - zg) * ng + zg * hold;
      h[o]  = hnew;
      hb[o] = (__bf16)hnew;
    }
  }
}

// ---------------- readout: out[b] = Σ_i h[b*1024+i] ----------------
__global__ void k_readout(const float* __restrict__ h, float* __restrict__ out) {
  int b = blockIdx.x;
  int c = threadIdx.x;
  float s = 0.0f;
  for (int i = 0; i < MAXNN; ++i) s += h[((size_t)b * MAXNN + i) * Hd + c];
  out[(size_t)b * Hd + c] = s;
}

// ---------------- host launch ----------------
extern "C" void kernel_launch(void* const* d_in, const int* in_sizes, int n_in,
                              void* d_out, int out_size, void* d_ws, size_t ws_size,
                              hipStream_t stream) {
  const float* X    = (const float*)d_in[0];
  const int*   ei   = (const int*)d_in[1];   // (2,E): [0..E)=src, [E..2E)=dst
  const int*   et   = (const int*)d_in[2];
  const float* Wp   = (const float*)d_in[3];
  const float* bp   = (const float*)d_in[4];
  const float* Wm   = (const float*)d_in[5];
  const float* bm   = (const float*)d_in[6];
  const float* Wih  = (const float*)d_in[7];
  const float* Whh  = (const float*)d_in[8];
  const float* bih  = (const float*)d_in[9];
  const float* bhh  = (const float*)d_in[10];
  float* out = (float*)d_out;

  char* ws = (char*)d_ws;
  size_t off = 0;
  auto carve = [&](size_t bytes) -> char* {
    char* p = ws + off;
    off = (off + bytes + 255) & ~(size_t)255;
    return p;
  };
  __bf16* Xb   = (__bf16*)carve((size_t)Nn * FP * 2);
  __bf16* Wpb  = (__bf16*)carve((size_t)Hd * FP * 2);
  __bf16* Wmb  = (__bf16*)carve((size_t)Ll * Hd * K2 * 2);
  __bf16* Wihb = (__bf16*)carve((size_t)Ll * G3 * Hd * 2);
  __bf16* Whhb = (__bf16*)carve((size_t)Ll * G3 * Hd * 2);
  float*  h    = (float*)carve((size_t)Nn * Hd * 4);
  __bf16* hb   = (__bf16*)carve((size_t)Nn * Hd * 2);
  __bf16* A    = (__bf16*)carve((size_t)Nn * K2 * 2);    // bf16 aggregate
  int*    cnt  = (int*)carve((size_t)Nn * Tt * 4);
  __bf16* mb   = (__bf16*)carve((size_t)Nn * Hd * 2);

  k_conv_x <<<1024, 256, 0, stream>>>(X, Xb);
  k_conv_wp<<<64,   256, 0, stream>>>(Wp, Wpb);
  k_conv_wm<<<1024, 256, 0, stream>>>(Wm, Wmb);
  k_conv_w2<<<512,  256, 0, stream>>>(Wih, Whh, Wihb, Whhb);

  k_proj<<<(Nn / 16) * (Hd / 16) / 8, 256, 0, stream>>>(Xb, Wpb, bp, h, hb);

  for (int l = 0; l < Ll; ++l) {
    hipMemsetAsync(A,   0, (size_t)Nn * K2 * 2, stream);
    hipMemsetAsync(cnt, 0, (size_t)Nn * Tt * 4, stream);
    k_edge<<<Ee / 8, 256, 0, stream>>>(hb, ei, ei + Ee, et, A, cnt);
    k_msg<<<Nn / 16, 256, 0, stream>>>(
        A, Wmb + (size_t)l * Hd * K2, cnt, bm + (size_t)l * Tt * Hd, mb);
    k_gru<<<Nn / 16, 256, 0, stream>>>(
        mb, hb, h, Wihb + (size_t)l * G3 * Hd, Whhb + (size_t)l * G3 * Hd,
        bih + (size_t)l * G3, bhh + (size_t)l * G3);
  }
  k_readout<<<Bb, Hd, 0, stream>>>(h, out);
  (void)in_sizes; (void)n_in; (void)out_size; (void)ws_size;
}
